// Conv2d_81106162417836
// MI455X (gfx1250) — matrix operand order repaired and checked by reading, not compiled, not run on hardware
//
#include <hip/hip_runtime.h>
#include <hip/hip_bf16.h>

typedef __attribute__((ext_vector_type(16))) _Float16 v16h;
typedef __attribute__((ext_vector_type(8)))  _Float16 v8h;
typedef __attribute__((ext_vector_type(4)))  _Float16 v4h;
typedef __attribute__((ext_vector_type(8)))  float    v8f;

#define CO      128       // output channels
#define CI      128       // input channels
#define HW      56
#define PIX_IMG (HW*HW)   // 3136
#define K_TOT   (CI*9)    // 1152
#define KC      32        // K chunk per iteration
#define NITER   (K_TOT/KC) // 36  (tap-major: tap = it>>2, ci_base = (it&3)*32)
#define LDA     40        // padded LDS stride (halves) for A
#define LDB     40        // padded LDS stride (halves) for B

static __device__ __forceinline__ int clampi(int v, int lo, int hi) {
    return v < lo ? lo : (v > hi ? hi : v);
}

// One-shot: W[o][ci][kh][kw] f32  ->  Wk[o][tap*128 + ci] f16 (k'-contiguous rows)
__global__ __launch_bounds__(256)
void transpose_weights_f16(const float* __restrict__ wgt,
                           _Float16* __restrict__ wk)
{
    const int idx = blockIdx.x * 256 + threadIdx.x;   // 0 .. 147455
    const int o   = idx / K_TOT;
    const int t   = idx - o * K_TOT;                  // tap*128 + ci
    const int tap = t >> 7;
    const int ci  = t & 127;
    wk[idx] = (_Float16)wgt[o * K_TOT + ci * 9 + tap];
}

template <bool PRETRANS>
__global__ __launch_bounds__(256, 2)
void conv2d_implicit_gemm_wmma(const float* __restrict__ x,
                               const float* __restrict__ wgt,
                               const _Float16* __restrict__ wk,
                               const float* __restrict__ bias,
                               float* __restrict__ out)
{
    // double-buffered tiles: A=[co][k'] f16, B=[pix][k'] f16
    __shared__ _Float16 Alds[2][CO * LDA]; // 2 x 10240 B
    __shared__ _Float16 Blds[2][32 * LDB]; // 2 x  2560 B

    const int tid  = threadIdx.x;
    const int lane = tid & 31;
    const int wave = tid >> 5;      // 0..7 -> co tile (16 rows each)
    const int g    = lane >> 4;     // lane half-group
    const int ln   = lane & 15;

    const unsigned P0 = blockIdx.x * 32;  // pixel tile base (exact cover)

    // ---- staging work assignment ----
    const int am  = tid >> 1;           // A: weight row (co) 0..127
    const int ak  = (tid & 1) * 16;     // A: k' offset within chunk (16 halves)
    const int bp  = tid & 31;           // B: pixel within tile
    const int bkg = (tid >> 5) * 4;     // B: ci offset within chunk (4 elems)

    // decode this thread's B pixel once
    const unsigned pg   = P0 + bp;
    const unsigned bn   = pg / (unsigned)PIX_IMG;
    const unsigned brem = pg - bn * (unsigned)PIX_IMG;
    const int      bh   = (int)(brem / (unsigned)HW);
    const int      bw   = (int)(brem - (unsigned)bh * (unsigned)HW);
    const float* __restrict__ xn =
        x + (size_t)bn * (size_t)(CI * PIX_IMG) + (size_t)bkg * PIX_IMG;

    v8f acc0 = {};
    v8f acc1 = {};

    for (int it = 0; it < NITER; ++it) {
        const int buf     = it & 1;
        const int tap     = it >> 2;         // uniform -> SALU
        const int ci_base = (it & 3) * 32;   // uniform -> SALU

        // ---- stage A: weights [128co x 32k'] -> f16 LDS, k'-contiguous ----
        if (PRETRANS) {
            const _Float16* __restrict__ src = wk + am * K_TOT + it * KC + ak;
            v8h h0 = *(const v8h*)(src);
            v8h h1 = *(const v8h*)(src + 8);
            *(v8h*)(&Alds[buf][am * LDA + ak])     = h0;
            *(v8h*)(&Alds[buf][am * LDA + ak + 8]) = h1;
        } else {
            // gather stride-9 from original layout: W[am][ci_base+ak+e][tap]
            const float* __restrict__ src = wgt + am * K_TOT + ci_base * 9 + tap;
            _Float16 hv[16];
            #pragma unroll
            for (int e = 0; e < 16; ++e)
                hv[e] = (_Float16)src[(ak + e) * 9];
            #pragma unroll
            for (int e = 0; e < 16; ++e)
                Alds[buf][am * LDA + ak + e] = hv[e];
        }

        // ---- stage B: im2col patch, tap uniform per iteration ----
        {
            const int kh = tap / 3;          // uniform
            const int kw = tap - kh * 3;     // uniform
            const int ih = bh + kh - 1;
            const int iw = bw + kw - 1;
            const bool inb = ((unsigned)ih < (unsigned)HW) &
                             ((unsigned)iw < (unsigned)HW);
            const int ihc = clampi(ih, 0, HW - 1);
            const int iwc = clampi(iw, 0, HW - 1);
            const float* __restrict__ bptr =
                xn + (size_t)ci_base * PIX_IMG + (unsigned)(ihc * HW + iwc);
            float bv[4];
            #pragma unroll
            for (int e = 0; e < 4; ++e) {
                const float v = bptr[e * PIX_IMG];   // immediate offsets
                bv[e] = inb ? v : 0.0f;              // one shared predicate
            }
            v4h hv;
            hv.x = (_Float16)bv[0];
            hv.y = (_Float16)bv[1];
            hv.z = (_Float16)bv[2];
            hv.w = (_Float16)bv[3];
            *(v4h*)(&Blds[buf][bp * LDB + bkg]) = hv;
        }

        __syncthreads();   // staging of buf done; next iter writes buf^1

        // ---- load fragments (k = g*16 + s in both A and B, same order) ----
        const int arow = wave * 16 + ln;
        v8h alo = *(const v8h*)(&Alds[buf][arow * LDA + g * 16]);
        v8h ahi = *(const v8h*)(&Alds[buf][arow * LDA + g * 16 + 8]);
        v16h a = __builtin_shufflevector(alo, ahi,
                                         0,1,2,3,4,5,6,7,8,9,10,11,12,13,14,15);

        v8h b0lo = *(const v8h*)(&Blds[buf][ln * LDB + g * 16]);
        v8h b0hi = *(const v8h*)(&Blds[buf][ln * LDB + g * 16 + 8]);
        v16h b0 = __builtin_shufflevector(b0lo, b0hi,
                                          0,1,2,3,4,5,6,7,8,9,10,11,12,13,14,15);

        v8h b1lo = *(const v8h*)(&Blds[buf][(16 + ln) * LDB + g * 16]);
        v8h b1hi = *(const v8h*)(&Blds[buf][(16 + ln) * LDB + g * 16 + 8]);
        v16h b1 = __builtin_shufflevector(b1lo, b1hi,
                                          0,1,2,3,4,5,6,7,8,9,10,11,12,13,14,15);

        // D = A x B + C  (f32 accumulate)
        acc0 = __builtin_amdgcn_wmma_f32_16x16x32_f16(
                   false, a, false, b0, (short)0, acc0, false, false);
        acc1 = __builtin_amdgcn_wmma_f32_16x16x32_f16(
                   false, a, false, b1, (short)0, acc1, false, false);
    }

    // ---- epilogue: D element r -> co = wave*16 + g*8 + r, pixel = ln (+16) ----
    {
        const unsigned pg0  = P0 + (unsigned)ln;
        const unsigned n0   = pg0 / (unsigned)PIX_IMG;
        const unsigned rem0 = pg0 - n0 * (unsigned)PIX_IMG;
        const size_t base0  = (size_t)n0 * (size_t)(CO * PIX_IMG) + rem0;

        const unsigned pg1  = P0 + 16u + (unsigned)ln;
        const unsigned n1   = pg1 / (unsigned)PIX_IMG;
        const unsigned rem1 = pg1 - n1 * (unsigned)PIX_IMG;
        const size_t base1  = (size_t)n1 * (size_t)(CO * PIX_IMG) + rem1;

        #pragma unroll
        for (int r = 0; r < 8; ++r) {
            const int co = wave * 16 + g * 8 + r;
            const float b = bias[co];
            out[base0 + (size_t)co * PIX_IMG] = acc0[r] + b;
            out[base1 + (size_t)co * PIX_IMG] = acc1[r] + b;
        }
    }
}

extern "C" void kernel_launch(void* const* d_in, const int* in_sizes, int n_in,
                              void* d_out, int out_size, void* d_ws, size_t ws_size,
                              hipStream_t stream) {
    (void)n_in; (void)out_size;
    const float* x    = (const float*)d_in[0];
    const float* wgt  = (const float*)d_in[1];
    const float* bias = (const float*)d_in[2];
    float* out = (float*)d_out;

    const int pixels = in_sizes[0] / CI;   // N*H*W = 100352
    const int blocks = pixels / 32;        // 3136 tiles, exact

    const size_t wk_bytes = (size_t)CO * K_TOT * sizeof(_Float16); // 294912 B
    if (ws_size >= wk_bytes) {
        _Float16* wk = (_Float16*)d_ws;
        transpose_weights_f16<<<(CO * K_TOT) / 256, 256, 0, stream>>>(wgt, wk);
        conv2d_implicit_gemm_wmma<true><<<blocks, 256, 0, stream>>>(
            x, wgt, wk, bias, out);
    } else {
        conv2d_implicit_gemm_wmma<false><<<blocks, 256, 0, stream>>>(
            x, wgt, (const _Float16*)nullptr, bias, out);
    }
}
